// SymmetricLinear_55679956025571
// MI455X (gfx1250) — matrix operand order repaired and checked by reading, not compiled, not run on hardware
//
#include <hip/hip_runtime.h>

// ---------------------------------------------------------------------------
// SymmetricLinear on MI455X (gfx1250, wave32, WMMA).
//
// Whole op == one GEMM:  Out(N x 576) = [x_iso | x_full](N x 576) @ BigW + bias
//   cols 0..63   : iso out   (W_ii  +  (W_fi/8 repeated over u))
//   cols 64..575 : full out  (W_if tiled over v  +  K[c, table[u,v], d])
// Roofline: 87 GFLOP vs 604 MB HBM traffic -> memory bound (~26us @ 23.3TB/s);
// bf16 WMMA keeps compute under the memory floor, f32 WMMA would not.
// ---------------------------------------------------------------------------

typedef __attribute__((ext_vector_type(16))) __bf16 v16bf;
typedef __attribute__((ext_vector_type(8)))  __bf16 v8bf;
typedef __attribute__((ext_vector_type(8)))  float  v8f;

#define M_TOTAL 131072     // B*H*W = 128*32*32
#define K_TOTAL 576        // 64 iso-in  + 8*64 full-in
#define N_TOTAL 576        // 64 iso-out + 8*64 full-out
#define BM      64
#define BN      64
#define LDA     584        // padded LDS row stride (bf16): 292 dwords, 292%64=36 -> conflict-free
#define KCHUNKS 18         // 576 / 32

__device__ __forceinline__ unsigned short f2bf(float f) {
    unsigned u = __float_as_uint(f);
    u += 0x7FFFu + ((u >> 16) & 1u);        // round-to-nearest-even
    return (unsigned short)(u >> 16);
}

// Build BigW^T (N_TOTAL x K_TOTAL bf16, row-major by OUTPUT column n, so each
// fragment's K-run is contiguous) and the fused bias vector (f32).
__global__ void symlin_prep_kernel(const float* __restrict__ Wii, const float* __restrict__ Wfi,
                                   const float* __restrict__ Wif, const float* __restrict__ Kt,
                                   const float* __restrict__ bii, const float* __restrict__ bif,
                                   unsigned short* __restrict__ wsB, float* __restrict__ wsBias) {
    int idx = blockIdx.x * blockDim.x + threadIdx.x;
    if (idx < N_TOTAL)
        wsBias[idx] = (idx < 64) ? bii[idx] : bif[(idx - 64) & 63];
    if (idx >= N_TOTAL * K_TOTAL) return;
    int n = idx / K_TOTAL;
    int k = idx - n * K_TOTAL;
    float val;
    if (n < 64) {                                   // iso output column d = n
        val = (k < 64) ? Wii[k * 64 + n]                       // iso2iso
                       : Wfi[((k - 64) & 63) * 64 + n] * 0.125f; // mean_r folded: W_fi/8
    } else {                                        // full output (v, d)
        int v = (n - 64) >> 6, d = (n - 64) & 63;
        if (k < 64) {
            val = Wif[k * 64 + d];                  // iso2full, broadcast over v
        } else {
            int u = (k - 64) >> 6, c = (k - 64) & 63;
            // D4 group table: table[u][v] = idx(mats[v]^{-1} @ mats[u])
            int g = ((u < 4) == (v < 4)) ? ((u - v) & 3) : (4 + ((u + v) & 3));
            val = Kt[c * 512 + g * 64 + d];         // K is (C_IN=64, R=8, C_OUT=64)
        }
    }
    wsB[idx] = f2bf(val);
}

// 64x64 output tile per block, full K=576 staged once in LDS, 8 waves:
// wave w -> M sub-tile (w&3), N half (w>>2) of two 16-wide tiles.
__global__ __launch_bounds__(256)
void symlin_gemm_kernel(const float* __restrict__ xiso, const float* __restrict__ xfull,
                        const unsigned short* __restrict__ wsB, const float* __restrict__ wsBias,
                        float* __restrict__ out) {
    __shared__ unsigned short sA[BM * LDA];   //  74,752 B
    __shared__ unsigned short sB[BN * LDA];   //  74,752 B   (total ~146 KB of 320 KB WGP pool)

    const int tid   = threadIdx.x;
    const int nBase = blockIdx.x * BN;        // N fastest -> 9 siblings share A tile in L2
    const int mBase = blockIdx.y * BM;

    // ---- stage A: f32 -> bf16, coalesced float4 reads --------------------
    {   // iso part: 64 rows x 64 cols = 1024 float4
        const float4* g = (const float4*)(xiso + (size_t)mBase * 64);
        #pragma unroll
        for (int i = 0; i < 4; ++i) {
            int q = tid + i * 256;
            int r = q >> 4, c4 = q & 15;
            float4 f = g[(size_t)r * 16 + c4];
            unsigned short* p = sA + r * LDA + c4 * 4;
            *(unsigned*)(p)     = (unsigned)f2bf(f.x) | ((unsigned)f2bf(f.y) << 16);
            *(unsigned*)(p + 2) = (unsigned)f2bf(f.z) | ((unsigned)f2bf(f.w) << 16);
        }
        // full part: 64 rows x 512 cols = 8192 float4
        const float4* gf = (const float4*)(xfull + (size_t)mBase * 512);
        #pragma unroll 4
        for (int i = 0; i < 32; ++i) {
            int q = tid + i * 256;
            int r = q >> 7, c4 = q & 127;
            float4 f = gf[(size_t)r * 128 + c4];
            unsigned short* p = sA + r * LDA + 64 + c4 * 4;
            *(unsigned*)(p)     = (unsigned)f2bf(f.x) | ((unsigned)f2bf(f.y) << 16);
            *(unsigned*)(p + 2) = (unsigned)f2bf(f.z) | ((unsigned)f2bf(f.w) << 16);
        }
    }
    // ---- stage B: already bf16 in workspace (L2-hot, 663 KB total) -------
    {
        int r = tid >> 2, part = tid & 3;     // 4 threads per B row, 72 8-elem chunks/row
        const unsigned short* gB = wsB + (size_t)(nBase + r) * K_TOTAL;
        unsigned short* lB = sB + r * LDA;
        #pragma unroll 6
        for (int j = 0; j < 18; ++j) {
            int ch = part + 4 * j;
            *(v8bf*)(lB + ch * 8) = *(const v8bf*)(gB + ch * 8);
        }
    }
    __syncthreads();

    // ---- compute: 18 chunks x 2 WMMA per wave ----------------------------
    const int lane = tid & 31;
    const int lm   = lane & 15;
    const int hi   = lane >> 4;
    const int mtl  = (tid >> 5) & 3;          // wave's M sub-tile
    const int nh   = tid >> 7;                // wave's N half (0/1)

    const unsigned short* aRow  = sA + (mtl * 16 + lm) * LDA;
    const unsigned short* bRow0 = sB + (nh * 32 + lm) * LDA;        // N tile 2*nh
    const unsigned short* bRow1 = sB + (nh * 32 + 16 + lm) * LDA;   // N tile 2*nh+1

    v8f acc0 = {};
    v8f acc1 = {};

    #pragma unroll
    for (int kc = 0; kc < KCHUNKS; ++kc) {
        const int k0 = kc * 32;
        v16bf a, b0, b1;
        // A 16x32 bf16 layout: lane row = lm, contiguous K = k0 + hi*16 .. +15
        ((v8bf*)&a)[0]  = *(const v8bf*)(aRow  + k0 + hi * 16);
        ((v8bf*)&a)[1]  = *(const v8bf*)(aRow  + k0 + hi * 16 + 8);
        // B 32x16 bf16 layout: lane col = lm, contiguous K = k0 + hi*16 .. +15
        ((v8bf*)&b0)[0] = *(const v8bf*)(bRow0 + k0 + hi * 16);
        ((v8bf*)&b0)[1] = *(const v8bf*)(bRow0 + k0 + hi * 16 + 8);
        ((v8bf*)&b1)[0] = *(const v8bf*)(bRow1 + k0 + hi * 16);
        ((v8bf*)&b1)[1] = *(const v8bf*)(bRow1 + k0 + hi * 16 + 8);
        acc0 = __builtin_amdgcn_wmma_f32_16x16x32_bf16(false, a, false, b0, (short)0, acc0, false, false);
        acc1 = __builtin_amdgcn_wmma_f32_16x16x32_bf16(false, a, false, b1, (short)0, acc1, false, false);
    }

    // ---- bias + store (C/D layout: col = lm, VGPR i -> row i + hi*8) -----
    const int col0 = nBase + nh * 32 + lm;
    const float bias0 = wsBias[col0];
    const float bias1 = wsBias[col0 + 16];
    const int rowg = mBase + mtl * 16 + hi * 8;
    float* o0 = out + (size_t)rowg * N_TOTAL + col0;
    #pragma unroll
    for (int i = 0; i < 8; ++i) {
        o0[(size_t)i * N_TOTAL]      = acc0[i] + bias0;
        o0[(size_t)i * N_TOTAL + 16] = acc1[i] + bias1;
    }
}

extern "C" void kernel_launch(void* const* d_in, const int* in_sizes, int n_in,
                              void* d_out, int out_size, void* d_ws, size_t ws_size,
                              hipStream_t stream) {
    const float* xiso  = (const float*)d_in[0];   // (131072, 64)
    const float* xfull = (const float*)d_in[1];   // (131072, 8, 64)
    const float* Wii   = (const float*)d_in[2];
    const float* bii   = (const float*)d_in[3];
    const float* Wfi   = (const float*)d_in[4];
    const float* Wif   = (const float*)d_in[5];
    const float* bif   = (const float*)d_in[6];
    const float* Kt    = (const float*)d_in[7];

    unsigned short* wsB   = (unsigned short*)d_ws;                       // 576*576 bf16 = 663,552 B
    float*          wsBias = (float*)((char*)d_ws + (size_t)N_TOTAL * K_TOTAL * 2);

    symlin_prep_kernel<<<dim3((N_TOTAL * K_TOTAL + 255) / 256), dim3(256), 0, stream>>>(
        Wii, Wfi, Wif, Kt, bii, bif, wsB, wsBias);

    symlin_gemm_kernel<<<dim3(N_TOTAL / BN, M_TOTAL / BM), dim3(256), 0, stream>>>(
        xiso, xfull, wsB, wsBias, (float*)d_out);
}
